// SelfAttention_10033043603510
// MI455X (gfx1250) — matrix operand order repaired and checked by reading, not compiled, not run on hardware
//
#include <hip/hip_runtime.h>
#include <hip/hip_bf16.h>

// ---------------------------------------------------------------------------
// Multi-head self-attention for MI455X (gfx1250), f16 WMMA / f32 accumulate.
// B=4, S=2048, E=1024, H=16, D=64.
// ---------------------------------------------------------------------------

typedef _Float16 half_t;
typedef __attribute__((ext_vector_type(16))) _Float16 v16h;
typedef __attribute__((ext_vector_type(8)))  _Float16 v8h;
typedef __attribute__((ext_vector_type(8)))  float    v8f;

#define EMBED 1024
#define NHEAD 16
#define HDIM  64
#define BATCH 4
#define SEQ   2048
#define ROWS  (BATCH * SEQ)     // 8192
#define KT    (EMBED / 32)      // 32  k-tiles in projection GEMMs
#define MT2   (ROWS / 32)       // 256 32-row tiles of X
#define QTILES (SEQ / 16)       // 128 query tiles per (b,h)
#define KVNT   (SEQ / 16)       // 128 key tiles (B-frag, 16 keys each)
#define KT2    (SEQ / 32)       // 64  key-pair tiles (32 keys per attn step)

static __device__ __forceinline__ v8f wmma_f16(v16h a, v16h b, v8f c) {
  // v_wmma_f32_16x16x32_f16  D = A(16x32,f16) * B(32x16,f16) + C(16x16,f32)
  return __builtin_amdgcn_wmma_f32_16x16x32_f16(false, a, false, b,
                                                (short)0, c, false, false);
}

// ---------------------------------------------------------------------------
// Fragment layouts (wave32), from CDNA5 ISA 7.12.2:
//   A (16x32 f16):  lane holds row M=lane%16; half h -> K = h%8 + (h/8)*16 + (lane/16)*8
//   B (32x16 f16):  lane holds col N=lane%16; half h -> K = h%8 + (h/8)*16 + (lane/16)*8
//   C (16x16 f32):  VGPR r, lane L -> M = r + 8*(L/16), N = L%16
// Fragments stored as [tile][lane][16 halves] -> contiguous 32B/lane loads.
// ---------------------------------------------------------------------------

__global__ void __launch_bounds__(256)
pack_w_kernel(const float* __restrict__ W, half_t* __restrict__ dst) {
  const int t    = blockIdx.x * 256 + threadIdx.x;  // 64*32*32 = 65536 slots
  const int g    = (t >> 4) & 1;
  const int ln   = t & 15;
  const int kt   = (t >> 5) & 31;
  const int nt   = t >> 10;
  const float* src = W + (size_t)(nt * 16 + ln) * EMBED + kt * 32 + g * 8;
  half_t* d = dst + (size_t)t * 16;
#pragma unroll
  for (int i = 0; i < 8; ++i) d[i] = (half_t)src[i];
#pragma unroll
  for (int i = 0; i < 8; ++i) d[8 + i] = (half_t)src[16 + i];
}

__global__ void __launch_bounds__(256)
pack_x_kernel(const float* __restrict__ X, half_t* __restrict__ dst) {
  const int t  = blockIdx.x * 256 + threadIdx.x;   // 512*32*32 = 524288 slots
  const int g  = (t >> 4) & 1;
  const int ln = t & 15;
  const int kt = (t >> 5) & 31;
  const int mt = t >> 10;
  const float* src = X + (size_t)(mt * 16 + ln) * EMBED + kt * 32 + g * 8;
  half_t* d = dst + (size_t)t * 16;
#pragma unroll
  for (int i = 0; i < 8; ++i) d[i] = (half_t)src[i];
#pragma unroll
  for (int i = 0; i < 8; ++i) d[8 + i] = (half_t)src[16 + i];
}

__global__ void __launch_bounds__(256)
pack_ctx_kernel(const half_t* __restrict__ X, half_t* __restrict__ dst) {
  const int t  = blockIdx.x * 256 + threadIdx.x;
  const int g  = (t >> 4) & 1;
  const int ln = t & 15;
  const int kt = (t >> 5) & 31;
  const int mt = t >> 10;
  const half_t* src = X + (size_t)(mt * 16 + ln) * EMBED + kt * 32 + g * 8;
  half_t* d = dst + (size_t)t * 16;
#pragma unroll
  for (int i = 0; i < 8; ++i) d[i] = src[i];
#pragma unroll
  for (int i = 0; i < 8; ++i) d[8 + i] = src[16 + i];
}

// per-head A-frags of Q: [bh][qt][kt(2)][lane][16]
__global__ void __launch_bounds__(256)
pack_q_kernel(const half_t* __restrict__ Q, half_t* __restrict__ dst) {
  const int t  = blockIdx.x * 256 + threadIdx.x;   // B*H*128*2*32 = 524288
  const int g  = (t >> 4) & 1;
  const int ln = t & 15;
  const int kt = (t >> 5) & 1;
  const int u  = t >> 6;
  const int qt = u & 127;
  const int h  = (u >> 7) & 15;
  const int b  = u >> 11;
  const half_t* src = Q + (size_t)(b * SEQ + qt * 16 + ln) * EMBED
                        + h * HDIM + kt * 32 + g * 8;
  half_t* d = dst + (size_t)t * 16;
#pragma unroll
  for (int i = 0; i < 8; ++i) d[i] = src[i];
#pragma unroll
  for (int i = 0; i < 8; ++i) d[8 + i] = src[16 + i];
}

// B-frags of K^T: [bh][nt][kt(2)][lane][16]
__global__ void __launch_bounds__(256)
pack_k_kernel(const half_t* __restrict__ K, half_t* __restrict__ dst) {
  const int t  = blockIdx.x * 256 + threadIdx.x;   // 524288
  const int g  = (t >> 4) & 1;
  const int ln = t & 15;
  const int kt = (t >> 5) & 1;
  const int u  = t >> 6;
  const int nt = u & 127;
  const int h  = (u >> 7) & 15;
  const int b  = u >> 11;
  const half_t* src = K + (size_t)(b * SEQ + nt * 16 + ln) * EMBED
                        + h * HDIM + kt * 32 + g * 8;
  half_t* d = dst + (size_t)t * 16;
#pragma unroll
  for (int i = 0; i < 8; ++i) d[i] = src[i];
#pragma unroll
  for (int i = 0; i < 8; ++i) d[8 + i] = src[16 + i];
}

// B-frags of V for P@V: [bh][kt2][nt(4)][lane][16]
__global__ void __launch_bounds__(256)
pack_v_kernel(const half_t* __restrict__ V, half_t* __restrict__ dst) {
  const int t   = blockIdx.x * 256 + threadIdx.x;  // B*H*64*4*32 = 524288
  const int g   = (t >> 4) & 1;
  const int ln  = t & 15;
  const int nt  = (t >> 5) & 3;
  const int u   = t >> 7;
  const int kt2 = u & 63;
  const int h   = (u >> 6) & 15;
  const int b   = u >> 10;
  const half_t* src = V + (size_t)(b * SEQ + kt2 * 32 + g * 8) * EMBED
                        + h * HDIM + nt * 16 + ln;
  half_t* d = dst + (size_t)t * 16;
#pragma unroll
  for (int i = 0; i < 16; ++i) d[i] = src[(size_t)(i + (i & 8)) * EMBED];
}

// ---------------------------------------------------------------------------
// GEMM: C[8192,1024] = A[8192,1024] @ W^T + bias, pre-packed fragments.
// One wave -> 32x64 output (2 A-frags x 4 B-frags, 8 accumulators).
// Software-pipelined: wrapping prefetch of next k-step's fragments so WMMAs
// overlap the in-flight global loads instead of s_wait_loadcnt 0 stalls.
// ---------------------------------------------------------------------------
__global__ void __launch_bounds__(256)
gemm16_kernel(const v16h* __restrict__ A, const v16h* __restrict__ Bp,
              const float* __restrict__ bias, half_t* __restrict__ outH,
              float* __restrict__ outF) {
  const int lane = threadIdx.x & 31;
  const int wid  = threadIdx.x >> 5;
  const int gw   = blockIdx.x * 8 + wid;   // 4096 waves = 256 mt2 * 16 ngroups
  const int mt2  = gw >> 4;
  const int ng   = gw & 15;
  const int g    = lane >> 4, ln = lane & 15;

  const v16h* a0 = A  + (size_t)(mt2 * 2 + 0) * KT * 32 + lane;
  const v16h* a1 = A  + (size_t)(mt2 * 2 + 1) * KT * 32 + lane;
  const v16h* b0 = Bp + (size_t)(ng * 4 + 0) * KT * 32 + lane;
  const v16h* b1 = Bp + (size_t)(ng * 4 + 1) * KT * 32 + lane;
  const v16h* b2 = Bp + (size_t)(ng * 4 + 2) * KT * 32 + lane;
  const v16h* b3 = Bp + (size_t)(ng * 4 + 3) * KT * 32 + lane;

  v8f c00{}, c01{}, c02{}, c03{};
  v8f c10{}, c11{}, c12{}, c13{};

  v16h a0c = a0[0], a1c = a1[0];
  v16h b0c = b0[0], b1c = b1[0], b2c = b2[0], b3c = b3[0];

#pragma unroll 2
  for (int kt = 0; kt < KT; ++kt) {
    const int kn = ((kt + 1) & (KT - 1)) * 32;   // wrap: branch-free prefetch
    const v16h a0n = a0[kn], a1n = a1[kn];
    const v16h b0n = b0[kn], b1n = b1[kn], b2n = b2[kn], b3n = b3[kn];

    c00 = wmma_f16(a0c, b0c, c00);
    c01 = wmma_f16(a0c, b1c, c01);
    c02 = wmma_f16(a0c, b2c, c02);
    c03 = wmma_f16(a0c, b3c, c03);
    c10 = wmma_f16(a1c, b0c, c10);
    c11 = wmma_f16(a1c, b1c, c11);
    c12 = wmma_f16(a1c, b2c, c12);
    c13 = wmma_f16(a1c, b3c, c13);

    a0c = a0n; a1c = a1n;
    b0c = b0n; b1c = b1n; b2c = b2n; b3c = b3n;
  }

  const int col = ng * 64 + ln;
  const float bb0 = bias[col],      bb1 = bias[col + 16];
  const float bb2 = bias[col + 32], bb3 = bias[col + 48];

#pragma unroll
  for (int r = 0; r < 8; ++r) {
    const int row0 = mt2 * 32 + r + 8 * g;
    const int row1 = row0 + 16;
    const size_t o0 = (size_t)row0 * EMBED + col;
    const size_t o1 = (size_t)row1 * EMBED + col;
    if (outH) {
      outH[o0]      = (half_t)(c00[r] + bb0);
      outH[o0 + 16] = (half_t)(c01[r] + bb1);
      outH[o0 + 32] = (half_t)(c02[r] + bb2);
      outH[o0 + 48] = (half_t)(c03[r] + bb3);
      outH[o1]      = (half_t)(c10[r] + bb0);
      outH[o1 + 16] = (half_t)(c11[r] + bb1);
      outH[o1 + 32] = (half_t)(c12[r] + bb2);
      outH[o1 + 48] = (half_t)(c13[r] + bb3);
    } else {
      outF[o0]      = c00[r] + bb0;
      outF[o0 + 16] = c01[r] + bb1;
      outF[o0 + 32] = c02[r] + bb2;
      outF[o0 + 48] = c03[r] + bb3;
      outF[o1]      = c10[r] + bb0;
      outF[o1 + 16] = c11[r] + bb1;
      outF[o1 + 32] = c12[r] + bb2;
      outF[o1 + 48] = c13[r] + bb3;
    }
  }
}

// ---------------------------------------------------------------------------
// Flash attention: one wave = 16 queries of one (b,h); 64 steps of 32 keys.
// V loads issued at step top (consumed after softmax); next-step K frags
// prefetched right after the score WMMAs -> loads hide under softmax VALU.
// ---------------------------------------------------------------------------
__global__ void __launch_bounds__(256)
attn_kernel(const v16h* __restrict__ Qf, const v16h* __restrict__ Kf,
            const v16h* __restrict__ Vf, const float* __restrict__ hscal,
            half_t* __restrict__ ctx) {
  // per-wave 16x32 f16 P tile; row stride 40 halves (80B) keeps the
  // 8-half A-frag segments 16B-aligned and spreads banks.
  __shared__ __align__(16) half_t plds[8][16 * 40];

  const int lane = threadIdx.x & 31;
  const int wid  = threadIdx.x >> 5;
  const int g    = lane >> 4, ln = lane & 15;
  const int gw   = blockIdx.x * 8 + wid;     // 8192 = 4*16*128
  const int qt   = gw & 127;
  const int h    = (gw >> 7) & 15;
  const int b    = gw >> 11;

  // head importance = softmax(head_scaling)[h]
  float hm = hscal[0];
#pragma unroll
  for (int i = 1; i < NHEAD; ++i) hm = fmaxf(hm, hscal[i]);
  float hsum = 0.f;
#pragma unroll
  for (int i = 0; i < NHEAD; ++i) hsum += __expf(hscal[i] - hm);
  const float himp = __expf(hscal[h] - hm) / hsum;

  const size_t bh = (size_t)(b * NHEAD + h);
  const v16h* qb = Qf + (bh * QTILES + qt) * 2 * 32 + lane;
  const v16h aq0 = qb[0];
  const v16h aq1 = qb[32];

  const v16h* kbase = Kf + (bh * KVNT) * 2 * 32 + lane;   // [nt][kt][lane]
  const v16h* vbase = Vf + (bh * KT2) * 4 * 32 + lane;    // [kt2][nt][lane]

  v8f o0{}, o1{}, o2{}, o3{};
  float m[8], l[8];
#pragma unroll
  for (int r = 0; r < 8; ++r) { m[r] = -1e30f; l[r] = 0.f; }

  half_t* my = &plds[wid][0];
  const float SCALE = 0.125f;        // 1/sqrt(64)
  const float L2E   = 1.44269504f;

  // prime K fragments for step 0
  v16h k0c = kbase[0], k1c = kbase[32], k2c = kbase[64], k3c = kbase[96];

  for (int it = 0; it < KT2; ++it) {
    // V frags for this step: issue early, consumed after the softmax
    const v16h* vb = vbase + (size_t)it * 4 * 32;
    const v16h v0 = vb[0], v1 = vb[32], v2 = vb[64], v3 = vb[96];

    v8f s0{}, s1{};
    s0 = wmma_f16(aq0, k0c, s0);
    s0 = wmma_f16(aq1, k1c, s0);
    s1 = wmma_f16(aq0, k2c, s1);
    s1 = wmma_f16(aq1, k3c, s1);

    // prefetch next step's K frags (wrapping, branch-free)
    const size_t kn = (size_t)((it + 1) & (KT2 - 1)) * 128;
    const v16h k0n = kbase[kn], k1n = kbase[kn + 32];
    const v16h k2n = kbase[kn + 64], k3n = kbase[kn + 96];

    float alpha[8];
#pragma unroll
    for (int r = 0; r < 8; ++r) {
      const float x0 = s0[r] * SCALE;
      const float x1 = s1[r] * SCALE;
      float rm = fmaxf(x0, x1);
      rm = fmaxf(rm, __shfl_xor(rm, 1));
      rm = fmaxf(rm, __shfl_xor(rm, 2));
      rm = fmaxf(rm, __shfl_xor(rm, 4));
      rm = fmaxf(rm, __shfl_xor(rm, 8));
      const float mn = fmaxf(m[r], rm);
      const float a  = exp2f((m[r] - mn) * L2E);
      const float p0 = exp2f((x0 - mn) * L2E);
      const float p1 = exp2f((x1 - mn) * L2E);
      float rs = p0 + p1;
      rs += __shfl_xor(rs, 1);
      rs += __shfl_xor(rs, 2);
      rs += __shfl_xor(rs, 4);
      rs += __shfl_xor(rs, 8);
      l[r] = l[r] * a + rs;
      m[r] = mn;
      alpha[r] = a;
      const int row = r + 8 * g;              // C-layout row of this VGPR
      my[row * 40 + ln]      = (half_t)p0;    // P[row][ 0..15]
      my[row * 40 + 16 + ln] = (half_t)p1;    // P[row][16..31]
    }
#pragma unroll
    for (int r = 0; r < 8; ++r) {
      o0[r] *= alpha[r]; o1[r] *= alpha[r];
      o2[r] *= alpha[r]; o3[r] *= alpha[r];
    }

    // LDS ops are in-order per wave; wait for store data before reload.
    asm volatile("s_wait_dscnt 0x0" ::: "memory");

    // re-read P as a 16x32 A-fragment (two b128 DS loads per lane)
    const v8h plo = *(const v8h*)&my[ln * 40 + g * 8];
    const v8h phi = *(const v8h*)&my[ln * 40 + 16 + g * 8];
    const v16h ap = __builtin_shufflevector(plo, phi,
        0, 1, 2, 3, 4, 5, 6, 7, 8, 9, 10, 11, 12, 13, 14, 15);

    o0 = wmma_f16(ap, v0, o0);
    o1 = wmma_f16(ap, v1, o1);
    o2 = wmma_f16(ap, v2, o2);
    o3 = wmma_f16(ap, v3, o3);

    k0c = k0n; k1c = k1n; k2c = k2n; k3c = k3n;
  }

#pragma unroll
  for (int r = 0; r < 8; ++r) {
    const float sc = himp / l[r];
    const int row = qt * 16 + r + 8 * g;
    const size_t o = ((size_t)b * SEQ + row) * EMBED + h * HDIM + ln;
    ctx[o]      = (half_t)(o0[r] * sc);
    ctx[o + 16] = (half_t)(o1[r] * sc);
    ctx[o + 32] = (half_t)(o2[r] * sc);
    ctx[o + 48] = (half_t)(o3[r] * sc);
  }
}

// ---------------------------------------------------------------------------
extern "C" void kernel_launch(void* const* d_in, const int* in_sizes, int n_in,
                              void* d_out, int out_size, void* d_ws,
                              size_t ws_size, hipStream_t stream) {
  (void)in_sizes; (void)n_in; (void)out_size; (void)ws_size;
  const float* x  = (const float*)d_in[0];
  const float* Wq = (const float*)d_in[1];
  const float* bq = (const float*)d_in[2];
  const float* Wk = (const float*)d_in[3];
  const float* bk = (const float*)d_in[4];
  const float* Wv = (const float*)d_in[5];
  const float* bv = (const float*)d_in[6];
  const float* Wo = (const float*)d_in[7];
  const float* bo = (const float*)d_in[8];
  const float* hs = (const float*)d_in[9];
  float* out = (float*)d_out;

  // workspace layout (bytes); total 120 MiB with aliasing
  const size_t MB = 1024 * 1024;
  char* ws = (char*)d_ws;
  half_t* wpack   = (half_t*)(ws + 0);        //  8 MiB: 4x packed weights
  half_t* xpack   = (half_t*)(ws + 8 * MB);   // 16 MiB (reused as ctxpack)
  half_t* qrm     = (half_t*)(ws + 24 * MB);  // 16 MiB (reused as ctxrm)
  half_t* krm     = (half_t*)(ws + 40 * MB);  // 16 MiB
  half_t* vrm     = (half_t*)(ws + 56 * MB);  // 16 MiB
  half_t* qfrag   = (half_t*)(ws + 72 * MB);  // 16 MiB
  half_t* kfrag   = (half_t*)(ws + 88 * MB);  // 16 MiB
  half_t* vfrag   = (half_t*)(ws + 104 * MB); // 16 MiB
  half_t* ctxrm   = qrm;
  half_t* ctxpack = xpack;
  const size_t WELEMS = (size_t)EMBED * EMBED;  // halves per packed weight

  // stage 0: pack weights + activations
  pack_w_kernel<<<256, 256, 0, stream>>>(Wq, wpack + 0 * WELEMS);
  pack_w_kernel<<<256, 256, 0, stream>>>(Wk, wpack + 1 * WELEMS);
  pack_w_kernel<<<256, 256, 0, stream>>>(Wv, wpack + 2 * WELEMS);
  pack_w_kernel<<<256, 256, 0, stream>>>(Wo, wpack + 3 * WELEMS);
  pack_x_kernel<<<2048, 256, 0, stream>>>(x, xpack);

  // stage 1: Q/K/V projections (f16 row-major outputs)
  gemm16_kernel<<<512, 256, 0, stream>>>((const v16h*)xpack,
      (const v16h*)(wpack + 0 * WELEMS), bq, qrm, nullptr);
  gemm16_kernel<<<512, 256, 0, stream>>>((const v16h*)xpack,
      (const v16h*)(wpack + 1 * WELEMS), bk, krm, nullptr);
  gemm16_kernel<<<512, 256, 0, stream>>>((const v16h*)xpack,
      (const v16h*)(wpack + 2 * WELEMS), bv, vrm, nullptr);

  // stage 2: repack per-head fragments, then flash attention
  pack_q_kernel<<<2048, 256, 0, stream>>>(qrm, qfrag);
  pack_k_kernel<<<2048, 256, 0, stream>>>(krm, kfrag);
  pack_v_kernel<<<2048, 256, 0, stream>>>(vrm, vfrag);
  attn_kernel<<<1024, 256, 0, stream>>>((const v16h*)qfrag,
      (const v16h*)kfrag, (const v16h*)vfrag, hs, ctxrm);

  // stage 3: output projection (f32 + bias straight to d_out)
  pack_ctx_kernel<<<2048, 256, 0, stream>>>(ctxrm, ctxpack);
  gemm16_kernel<<<512, 256, 0, stream>>>((const v16h*)ctxpack,
      (const v16h*)(wpack + 3 * WELEMS), bo, nullptr, out);
}
